// GraphAttentionLayer_21620865368515
// MI455X (gfx1250) — matrix operand order repaired and checked by reading, not compiled, not run on hardware
//
#include <hip/hip_runtime.h>
#include <hip/hip_bf16.h>

typedef __attribute__((ext_vector_type(16))) _Float16 v16h;
typedef __attribute__((ext_vector_type(8)))  _Float16 v8h;
typedef __attribute__((ext_vector_type(8)))  float    v8f;

#define GAT_N    4096
#define GAT_K    512
#define GAT_DO   256
#define NEG_BIG  (-9.0e15f)
#define PSTR     48   // padded LDS stride (halves) for [col][k] panels: 96B rows, 32B-aligned frags
#define ASTR     36   // padded LDS stride (ints) for adj tile: 144B rows, 16B-aligned int4s

union H2U { _Float16 h[2]; unsigned u; };

// ---------------------------------------------------------------------------
// Kernel 1: Wh = h @ Ws  (B x 4096x512 * 512x256), f16 WMMA, f32 accumulate.
// LDS panels stored transposed [col][k] so fragments load as contiguous b128s.
// ---------------------------------------------------------------------------
struct SmemG1 {
    _Float16 hA[32 * PSTR];    // [row_local][k]   (transposed-friendly: row-major, k contiguous)
    _Float16 wB[256 * PSTR];   // [col][k]         (transposed)
};

__global__ __launch_bounds__(256) void gat_gemm1(
    const float* __restrict__ h, const float* __restrict__ Ws,
    float* __restrict__ Wh, _Float16* __restrict__ WhH)
{
    __shared__ SmemG1 sm;
    const int b       = blockIdx.y;
    const int rowTile = blockIdx.x * 32;
    const int t       = threadIdx.x;
    const int lane    = t & 31;
    const int wave    = t >> 5;
    const int rowgrp  = wave >> 2;
    const int colgrp  = wave & 3;
    const int hi      = (lane >> 4) & 1;

    v8f acc[4];
#pragma unroll
    for (int n = 0; n < 4; ++n)
#pragma unroll
        for (int v = 0; v < 8; ++v) acc[n][v] = 0.0f;

    for (int kc = 0; kc < GAT_K / 32; ++kc) {
        const int k0 = kc * 32;
        // --- stage h tile 32x32 (f32 -> f16), [row][k], k contiguous ---
        {
            const int il = t >> 3, kl4 = (t & 7) * 4;
            float4 hv = *(const float4*)(h + ((size_t)(b * GAT_N + rowTile + il)) * GAT_K + k0 + kl4);
            H2U p01, p23;
            p01.h[0] = (_Float16)hv.x; p01.h[1] = (_Float16)hv.y;
            p23.h[0] = (_Float16)hv.z; p23.h[1] = (_Float16)hv.w;
            *(unsigned*)&sm.hA[il * PSTR + kl4 + 0] = p01.u;
            *(unsigned*)&sm.hA[il * PSTR + kl4 + 2] = p23.u;
        }
        // --- stage Ws tile 32x256 (f32 -> f16), TRANSPOSED to [col][k] ---
        {
            const int kp = t >> 4, kl = kp * 2, cseg = (t & 15) * 16;
            const float* g0 = Ws + ((size_t)(b * GAT_K + k0 + kl)) * GAT_DO + cseg;
            float r0[16], r1[16];
#pragma unroll
            for (int q = 0; q < 4; ++q) {
                *(float4*)(r0 + q * 4) = *(const float4*)(g0 + q * 4);
                *(float4*)(r1 + q * 4) = *(const float4*)(g0 + GAT_DO + q * 4);
            }
#pragma unroll
            for (int c = 0; c < 16; ++c) {
                H2U p; p.h[0] = (_Float16)r0[c]; p.h[1] = (_Float16)r1[c];
                *(unsigned*)&sm.wB[(cseg + c) * PSTR + kl] = p.u;
            }
        }
        __syncthreads();

        // --- A fragment: two contiguous 8-half groups (k: hi*16.., hi*16+8..) ---
        const int arow = rowgrp * 16 + (lane & 15);
        v8h alo = *(const v8h*)(sm.hA + arow * PSTR + (hi << 4));
        v8h ahi = *(const v8h*)(sm.hA + arow * PSTR + 8 + (hi << 4));
        v16h af = __builtin_shufflevector(alo, ahi, 0,1,2,3,4,5,6,7,8,9,10,11,12,13,14,15);

        // --- B fragments: one contiguous v16h (2x ds_load_b128) each ---
        const int kb = hi << 4;
#pragma unroll
        for (int n = 0; n < 4; ++n) {
            const int col = colgrp * 64 + n * 16 + (lane & 15);
            v16h bf = *(const v16h*)(sm.wB + col * PSTR + kb);
            acc[n] = __builtin_amdgcn_wmma_f32_16x16x32_f16(
                false, af, false, bf, (short)0, acc[n], false, false);
        }
        __syncthreads();
    }

#pragma unroll
    for (int v = 0; v < 8; ++v) {
        const int row = rowTile + rowgrp * 16 + v + (hi << 3);
#pragma unroll
        for (int n = 0; n < 4; ++n) {
            const int col = colgrp * 64 + n * 16 + (lane & 15);
            const size_t idx = ((size_t)(b * GAT_N + row)) * GAT_DO + col;
            const float val = acc[n][v];
            Wh[idx]  = val;
            WhH[idx] = (_Float16)val;
        }
    }
}

// ---------------------------------------------------------------------------
// Kernel 2: e1/e2 = Wh · a1 / a2, one wave32 per row, shuffle reduction.
// ---------------------------------------------------------------------------
__global__ __launch_bounds__(256) void gat_e(
    const float* __restrict__ Wh, const float* __restrict__ a,
    float* __restrict__ e1, float* __restrict__ e2)
{
    const int gw   = blockIdx.x * 8 + (threadIdx.x >> 5);
    const int lane = threadIdx.x & 31;
    const int b    = gw >> 12;
    const float* wrow = Wh + (size_t)gw * GAT_DO;
    const float* ab   = a + b * (2 * GAT_DO);
    float s1 = 0.0f, s2 = 0.0f;
#pragma unroll
    for (int c = lane; c < GAT_DO; c += 32) {
        const float w = wrow[c];
        s1 += w * ab[c];
        s2 += w * ab[GAT_DO + c];
    }
#pragma unroll
    for (int m = 16; m >= 1; m >>= 1) {
        s1 += __shfl_xor(s1, m, 32);
        s2 += __shfl_xor(s2, m, 32);
    }
    if (lane == 0) { e1[gw] = s1; e2[gw] = s2; }
}

// ---------------------------------------------------------------------------
// Kernel 3: fused masked-softmax flash attention + ELU.
// Wh panel staged transposed [col][j] so B fragments are contiguous v16h.
// ---------------------------------------------------------------------------
struct SmemAtt {
    _Float16 wh[256 * PSTR];   // [col][j_local]  (~24 KB)
    int      adj[32 * ASTR];   // [i_local][j_local], padded (~4.5 KB)
    float    e2[32];
};

__global__ __launch_bounds__(256) void gat_attn(
    const float* __restrict__ e1g, const float* __restrict__ e2g,
    const int* __restrict__ adjg, const _Float16* __restrict__ WhH,
    float* __restrict__ out)
{
    __shared__ SmemAtt sm;
    const int b       = blockIdx.y;
    const int rowTile = blockIdx.x * 32;
    const int t       = threadIdx.x;
    const int lane    = t & 31;
    const int wave    = t >> 5;
    const int rowgrp  = wave >> 2;
    const int colgrp  = wave & 3;
    const int hi      = (lane >> 4) & 1;
    const int r0      = rowTile + rowgrp * 16;

    const float e1r = e1g[b * GAT_N + r0 + (lane & 15)];

    float m = -3.0e38f, l = 0.0f;
    v8f acc[4];
#pragma unroll
    for (int n = 0; n < 4; ++n)
#pragma unroll
        for (int v = 0; v < 8; ++v) acc[n][v] = 0.0f;

    for (int jc = 0; jc < GAT_N / 32; ++jc) {
        const int j0 = jc * 32;
        // --- stage Wh panel 32x256 f16, TRANSPOSED to [col][j]; adj; e2 ---
        {
            const int jp = t >> 4, jl = jp * 2, cseg = (t & 15) * 16;
            const _Float16* g0 = WhH + ((size_t)(b * GAT_N + j0 + jl)) * GAT_DO + cseg;
            unsigned short c0[16], c1[16];
            *(uint4*)(c0 + 0) = *(const uint4*)(g0 + 0);
            *(uint4*)(c0 + 8) = *(const uint4*)(g0 + 8);
            *(uint4*)(c1 + 0) = *(const uint4*)(g0 + GAT_DO + 0);
            *(uint4*)(c1 + 8) = *(const uint4*)(g0 + GAT_DO + 8);
#pragma unroll
            for (int c = 0; c < 16; ++c) {
                unsigned v = (unsigned)c0[c] | ((unsigned)c1[c] << 16);
                *(unsigned*)&sm.wh[(cseg + c) * PSTR + jl] = v;
            }
            if (jc + 1 < GAT_N / 32)
                __builtin_prefetch((const char*)g0 + 32 * GAT_DO * sizeof(_Float16), 0, 1);

            const int il = t >> 3, j4 = (t & 7) * 4;
            *(int4*)(sm.adj + il * ASTR + j4) =
                *(const int4*)(adjg + (size_t)(rowTile + il) * GAT_N + j0 + j4);
            if (t < 32) sm.e2[t] = e2g[b * GAT_N + j0 + t];
        }
        __syncthreads();

        // --- scores in A-fragment element order ---
        float s[16];
        float cmax = -3.0e38f;
        const int arow = rowgrp * 16 + (lane & 15);
#pragma unroll
        for (int e = 0; e < 16; ++e) {
            const int jl = e + (hi << 4);
            const int av = sm.adj[arow * ASTR + jl];
            float sc = e1r + sm.e2[jl];
            sc = sc > 0.0f ? sc : 0.2f * sc;
            sc = av > 0 ? sc : NEG_BIG;
            s[e] = sc;
            cmax = fmaxf(cmax, sc);
        }
        cmax = fmaxf(cmax, __shfl_xor(cmax, 16, 32));
        const float mnew   = fmaxf(m, cmax);
        const float fscale = __expf(m - mnew);

        v16h af;
        float psum = 0.0f;
#pragma unroll
        for (int e = 0; e < 16; ++e) {
            const float p = (s[e] > -8.0e15f) ? __expf(s[e] - mnew) : 0.0f;
            psum += p;
            af[e] = (_Float16)p;
        }
        psum += __shfl_xor(psum, 16, 32);
        l = l * fscale + psum;
        m = mnew;

        // --- rescale accumulators only when some row max moved ---
        if (__any(fscale != 1.0f)) {
#pragma unroll
            for (int v = 0; v < 8; ++v) {
                const float fv = __shfl(fscale, v + (hi << 3), 32);
#pragma unroll
                for (int n = 0; n < 4; ++n) acc[n][v] *= fv;
            }
        }

        // --- B fragments (contiguous v16h) + WMMA ---
        const int kb = hi << 4;
#pragma unroll
        for (int n = 0; n < 4; ++n) {
            const int col = colgrp * 64 + n * 16 + (lane & 15);
            v16h bf = *(const v16h*)(sm.wh + col * PSTR + kb);
            acc[n] = __builtin_amdgcn_wmma_f32_16x16x32_f16(
                false, af, false, bf, (short)0, acc[n], false, false);
        }
        __syncthreads();
    }

    // --- epilogue: normalize, ELU, store ---
#pragma unroll
    for (int v = 0; v < 8; ++v) {
        const float lv  = __shfl(l, v + (hi << 3), 32);
        const float inv = 1.0f / lv;
        const int   row = r0 + v + (hi << 3);
#pragma unroll
        for (int n = 0; n < 4; ++n) {
            float o = acc[n][v] * inv;
            o = o > 0.0f ? o : (__expf(o) - 1.0f);
            out[((size_t)(b * GAT_N + row)) * GAT_DO + colgrp * 64 + n * 16 + (lane & 15)] = o;
        }
    }
}

// ---------------------------------------------------------------------------
extern "C" void kernel_launch(void* const* d_in, const int* in_sizes, int n_in,
                              void* d_out, int out_size, void* d_ws, size_t ws_size,
                              hipStream_t stream) {
    (void)in_sizes; (void)n_in; (void)out_size; (void)ws_size;
    const float* h   = (const float*)d_in[0];   // (B,N,512)
    const int*   adj = (const int*)  d_in[1];   // (N,N)
    const float* Ws  = (const float*)d_in[2];   // (B,512,256)
    const float* a   = (const float*)d_in[3];   // (B,512,1)
    float* out = (float*)d_out;

    char* ws = (char*)d_ws;
    float*    Wh  = (float*)ws;                                        // 16 MB
    _Float16* WhH = (_Float16*)(ws + (size_t)16 * 1024 * 1024);        //  8 MB
    float*    e1  = (float*)(ws + (size_t)24 * 1024 * 1024);           // 64 KB
    float*    e2  = e1 + 4 * GAT_N;                                    // 64 KB

    dim3 blk(256);
    gat_gemm1<<<dim3(GAT_N / 32, 4), blk, 0, stream>>>(h, Ws, Wh, WhH);
    gat_e    <<<dim3(4 * GAT_N / 8), blk, 0, stream>>>(Wh, a, e1, e2);
    gat_attn <<<dim3(GAT_N / 32, 4), blk, 0, stream>>>(e1, e2, adj, WhH, out);
}
